// Organism_31834297598712
// MI455X (gfx1250) — matrix operand order repaired and checked by reading, not compiled, not run on hardware
//
#include <hip/hip_runtime.h>
#include <math.h>

// ---------------- problem constants ----------------
// B=64, T=512, D=256, H=1024, L=256, O=256
// z = concat(x_t[256], s[256], h[1024])  -> K = 1536, gates N = 4096
#define BATCH   64
#define SEQT    512
#define HID     1024
#define LAT     256
#define NG      4096          // 256 n-tiles of 16

typedef __attribute__((ext_vector_type(16))) __bf16 v16bf;
typedef __attribute__((ext_vector_type(8)))  float  v8f;

// ---------------- WMMA fragment packing (CDNA5 ISA 7.12.2) ----------------
// A 16x32 bf16 tile, element (m, k):
//   kl=k%32; lane = ((kl&15)>>3)*16 + (m&15); e = (kl>>4)*8 + (kl&7)
// B 32x16 bf16 tile, element (k, n):
//   kl=k%32; lane = ((kl&15)>>3)*16 + (n&15); e = (kl>>4)*8 + (kl&7)
// Each tile = 32 lanes * 16 bf16 = 512 elements (1024 B), contiguous.

__device__ __forceinline__ int a_off(int m, int kl) {
  int lane = (((kl & 15) >> 3) << 4) + (m & 15);
  int e    = ((kl >> 4) << 3) + (kl & 7);
  return (lane << 4) + e;
}
__device__ __forceinline__ int b_off(int k, int n) {
  int lane = (((k & 15) >> 3) << 4) + (n & 15);
  return (lane << 4) + (((k & 31) >> 4) << 3) + (k & 7);
}
__device__ __forceinline__ float sigm(float x) { return 1.0f / (1.0f + __expf(-x)); }

// ---------------- prep kernels (run once per launch) ----------------

__global__ __launch_bounds__(256) void pack_wcat(const float* __restrict__ Wih,
                                                 const float* __restrict__ Whh,
                                                 __bf16* __restrict__ dst) {
  int idx = blockIdx.x * 256 + threadIdx.x;            // < 1536*4096
  int k = idx >> 12, n = idx & 4095;
  float v = (k < 512) ? Wih[(size_t)n * 512 + k]
                      : Whh[(size_t)n * 1024 + (k - 512)];
  dst[(size_t)((k >> 5) * 256 + (n >> 4)) * 512 + b_off(k, n)] = (__bf16)v;
}

// generic [N,K] row-major weight -> B-fragment tiles (kt*(N/16)+nt)
__global__ __launch_bounds__(256) void pack_b(const float* __restrict__ W,
                                              __bf16* __restrict__ dst,
                                              int K, int N) {
  int idx = blockIdx.x * 256 + threadIdx.x;
  if (idx >= K * N) return;
  int k = idx / N, n = idx % N;
  float v = W[(size_t)n * K + k];
  dst[(size_t)((k >> 5) * (N >> 4) + (n >> 4)) * 512 + b_off(k, n)] = (__bf16)v;
}

// x [B,T,D] fp32 -> per-t A-fragment tiles: xA + t*16384 elems, tile (d/32)*4 + m/16
__global__ __launch_bounds__(256) void pack_x(const float* __restrict__ x,
                                              __bf16* __restrict__ xA) {
  int idx = blockIdx.x * 256 + threadIdx.x;            // < 64*512*256
  int m = idx >> 17, rem = idx & 131071, t = rem >> 8, d = rem & 255;
  float v = x[idx];
  xA[(size_t)t * 16384 + (size_t)(((d >> 5) << 2) + (m >> 4)) * 512 + a_off(m, d & 31)]
      = (__bf16)v;
}

__global__ __launch_bounds__(256) void init_state(const float* __restrict__ bih,
                                                  const float* __restrict__ bhh,
                                                  float* __restrict__ biasg,
                                                  __bf16* __restrict__ zA,
                                                  float* __restrict__ c) {
  int idx = blockIdx.x * 256 + threadIdx.x;            // grid 384 -> 98304
  if (idx < 48 * 4 * 512) zA[idx] = (__bf16)0.0f;
  if (idx < BATCH * HID)  c[idx]  = 0.0f;
  if (idx < 4 * HID)      biasg[idx] = bih[idx] + bhh[idx];
}

// ---------------- per-step kernels ----------------

// gates = concat(x_t, s, h) @ Wcat^T + (b_ih+b_hh)
// grid 64 x 256thr (8 waves). Each wave owns TWO adjacent 16x16 C tiles
// (nt0 = 2*ng, nt1 = nt0+1) sharing one A fragment per k-step:
// per k-step: 1 A load + 2 B loads + 2 WMMAs  (1.5 fragment loads / WMMA).
__global__ __launch_bounds__(256) void gates_kernel(const __bf16* __restrict__ xAt,
                                                    const __bf16* __restrict__ zA,
                                                    const __bf16* __restrict__ WB,
                                                    const float* __restrict__ biasg,
                                                    float* __restrict__ gates) {
  const int wave = threadIdx.x >> 5, lane = threadIdx.x & 31;
  const int mt = wave & 3;
  const int ng = (blockIdx.x << 1) + (wave >> 2);      // 0..127
  const int nt0 = ng << 1;                             // 0..254 (even)
  const int lx = lane << 4;

  v8f acc0, acc1;
  const float bv0 = biasg[(nt0 << 4) + (lane & 15)];
  const float bv1 = biasg[((nt0 + 1) << 4) + (lane & 15)];
#pragma unroll
  for (int v = 0; v < 8; ++v) { acc0[v] = bv0; acc1[v] = bv1; }

  // B tiles (kt*256 + nt0) and (+1) are adjacent 1KB blocks
  const __bf16* wp = WB + ((size_t)nt0 << 9) + lx;     // advance by 256*512 per kt
  // x section: k-tiles 0..7 from per-step xA
#pragma unroll 2
  for (int kt = 0; kt < 8; ++kt) {
    v16bf a  = *(const v16bf*)(xAt + (size_t)(((kt << 2) + mt) << 9) + lx);
    v16bf b0 = *(const v16bf*)(wp + (size_t)kt * (256 * 512));
    v16bf b1 = *(const v16bf*)(wp + (size_t)kt * (256 * 512) + 512);
    acc0 = __builtin_amdgcn_wmma_f32_16x16x32_bf16(false, a, false, b0,
                                                   (short)0, acc0, false, false);
    acc1 = __builtin_amdgcn_wmma_f32_16x16x32_bf16(false, a, false, b1,
                                                   (short)0, acc1, false, false);
  }
  // s (kt 8..15) + h (kt 16..47) sections from recurrent fragment buffer
#pragma unroll 4
  for (int kt = 8; kt < 48; ++kt) {
    v16bf a  = *(const v16bf*)(zA + (size_t)(((kt << 2) + mt) << 9) + lx);
    v16bf b0 = *(const v16bf*)(wp + (size_t)kt * (256 * 512));
    v16bf b1 = *(const v16bf*)(wp + (size_t)kt * (256 * 512) + 512);
    acc0 = __builtin_amdgcn_wmma_f32_16x16x32_bf16(false, a, false, b0,
                                                   (short)0, acc0, false, false);
    acc1 = __builtin_amdgcn_wmma_f32_16x16x32_bf16(false, a, false, b1,
                                                   (short)0, acc1, false, false);
  }
  // C layout: m = v + 8*(lane>>4) (+16*mt), n = lane&15 (+16*nt)
  const int m0 = (mt << 4) + ((lane >> 4) << 3);
  const int n0 = (nt0 << 4) + (lane & 15);
#pragma unroll
  for (int v = 0; v < 8; ++v) {
    gates[(size_t)(m0 + v) * NG + n0]      = acc0[v];
    gates[(size_t)(m0 + v) * NG + n0 + 16] = acc1[v];
  }
}

// LSTM elementwise; writes h_new as bf16 A-fragments into zA (k = 512 + j)
__global__ __launch_bounds__(256) void lstm_ew(const float* __restrict__ g,
                                               float* __restrict__ c,
                                               __bf16* __restrict__ zA) {
  int idx = blockIdx.x * 256 + threadIdx.x;            // 64*1024
  int m = idx >> 10, j = idx & 1023;
  const float* gr = g + ((size_t)m << 12);
  float iv = sigm(gr[j]);
  float fv = sigm(gr[1024 + j]);
  float gv = tanhf(gr[2048 + j]);
  float ov = sigm(gr[3072 + j]);
  float cn = fv * c[idx] + iv * gv;
  c[idx] = cn;
  float h = ov * tanhf(cn);
  int k = 512 + j, kt = k >> 5;
  zA[(size_t)((kt << 2) + (m >> 4)) * 512 + a_off(m, k & 31)] = (__bf16)h;
}

// s_curr = tanh(h @ W_enc^T + b_enc); writes self_states and s fragments (kt 8..15)
__global__ __launch_bounds__(256) void enc_kernel(__bf16* __restrict__ zA,
                                                  const __bf16* __restrict__ WB,
                                                  const float* __restrict__ benc,
                                                  float* __restrict__ out, int t) {
  const int wave = threadIdx.x >> 5, lane = threadIdx.x & 31;
  const int w = blockIdx.x * 8 + wave;                 // 64 waves total
  const int mt = w & 3, nt = w >> 2, lx = lane << 4;

  v8f acc;
  const float bv = benc[(nt << 4) + (lane & 15)];
#pragma unroll
  for (int v = 0; v < 8; ++v) acc[v] = bv;

#pragma unroll 4
  for (int ktl = 0; ktl < 32; ++ktl) {
    int kt = 16 + ktl;                                 // h section of zA
    v16bf a = *(const v16bf*)(zA + (size_t)(((kt << 2) + mt) << 9) + lx);
    v16bf b = *(const v16bf*)(WB + (size_t)(((ktl << 4) + nt) << 9) + lx);
    acc = __builtin_amdgcn_wmma_f32_16x16x32_bf16(false, a, false, b,
                                                  (short)0, acc, false, false);
  }
  const size_t SS = (size_t)BATCH * SEQT * 256;        // self_states base
#pragma unroll
  for (int v = 0; v < 8; ++v) {
    int m = (mt << 4) + v + ((lane >> 4) << 3);
    int n = (nt << 4) + (lane & 15);
    float s = tanhf(acc[v]);
    out[SS + ((size_t)m * SEQT + t) * LAT + n] = s;
    int k = 256 + n, kt = k >> 5;                      // s section of zA
    zA[(size_t)((kt << 2) + (m >> 4)) * 512 + a_off(m, k & 31)] = (__bf16)s;
  }
}

// s_pred = tanh(s @ W_pred^T + b_pred); y = s @ W_act^T + b_act
__global__ __launch_bounds__(256) void pred_act_kernel(const __bf16* __restrict__ zA,
                                                       const __bf16* __restrict__ WpB,
                                                       const __bf16* __restrict__ WaB,
                                                       const float* __restrict__ bpred,
                                                       const float* __restrict__ bact,
                                                       float* __restrict__ out, int t) {
  const int wave = threadIdx.x >> 5, lane = threadIdx.x & 31;
  const int which = blockIdx.x >> 3;                   // 0 = pred, 1 = act
  const int w = ((blockIdx.x & 7) << 3) + wave;
  const int mt = w & 3, nt = w >> 2, lx = lane << 4;

  const __bf16* WB = which ? WaB : WpB;
  const float*  bb = which ? bact : bpred;

  v8f acc;
  const float bv = bb[(nt << 4) + (lane & 15)];
#pragma unroll
  for (int v = 0; v < 8; ++v) acc[v] = bv;

#pragma unroll
  for (int ktl = 0; ktl < 8; ++ktl) {
    int kt = 8 + ktl;                                  // s section of zA
    v16bf a = *(const v16bf*)(zA + (size_t)(((kt << 2) + mt) << 9) + lx);
    v16bf b = *(const v16bf*)(WB + (size_t)(((ktl << 4) + nt) << 9) + lx);
    acc = __builtin_amdgcn_wmma_f32_16x16x32_bf16(false, a, false, b,
                                                  (short)0, acc, false, false);
  }
  const size_t SP = (size_t)BATCH * SEQT * 512;        // self_predictions base
#pragma unroll
  for (int v = 0; v < 8; ++v) {
    int m = (mt << 4) + v + ((lane >> 4) << 3);
    int n = (nt << 4) + (lane & 15);
    float r = acc[v];
    size_t off;
    if (which) { off = ((size_t)m * SEQT + t) * 256 + n; }          // outputs
    else       { r = tanhf(r); off = SP + ((size_t)m * SEQT + t) * LAT + n; }
    out[off] = r;
  }
}

// ---------------- workspace layout (bytes) ----------------
#define OFF_WCAT   0u            // 48*256 tiles * 1024B = 12,582,912
#define OFF_WENC   12582912u     // 32*16*1024 = 524,288
#define OFF_WPRED  13107200u     // 8*16*1024  = 131,072
#define OFF_WACT   13238272u     // 131,072
#define OFF_BIASG  13369344u     // 16,384
#define OFF_XA     13385728u     // 512 * 32KB = 16,777,216
#define OFF_ZA     30162944u     // 48*4*1024  = 196,608
#define OFF_GATES  30359552u     // 64*4096*4  = 1,048,576
#define OFF_C      31408128u     // 64*1024*4  = 262,144  (end ~31.7 MB)

extern "C" void kernel_launch(void* const* d_in, const int* in_sizes, int n_in,
                              void* d_out, int out_size, void* d_ws, size_t ws_size,
                              hipStream_t stream) {
  const float* x     = (const float*)d_in[0];
  const float* Wih   = (const float*)d_in[1];
  const float* Whh   = (const float*)d_in[2];
  const float* bih   = (const float*)d_in[3];
  const float* bhh   = (const float*)d_in[4];
  const float* Wenc  = (const float*)d_in[5];
  const float* benc  = (const float*)d_in[6];
  const float* Wpred = (const float*)d_in[7];
  const float* bpred = (const float*)d_in[8];
  const float* Wact  = (const float*)d_in[9];
  const float* bact  = (const float*)d_in[10];
  float* out = (float*)d_out;

  char* ws = (char*)d_ws;
  __bf16* WcatB  = (__bf16*)(ws + OFF_WCAT);
  __bf16* WencB  = (__bf16*)(ws + OFF_WENC);
  __bf16* WpredB = (__bf16*)(ws + OFF_WPRED);
  __bf16* WactB  = (__bf16*)(ws + OFF_WACT);
  float*  biasg  = (float*) (ws + OFF_BIASG);
  __bf16* xA     = (__bf16*)(ws + OFF_XA);
  __bf16* zA     = (__bf16*)(ws + OFF_ZA);
  float*  gates  = (float*) (ws + OFF_GATES);
  float*  cstate = (float*) (ws + OFF_C);

  // one-time prep (re-run every call: deterministic)
  pack_wcat<<<24576, 256, 0, stream>>>(Wih, Whh, WcatB);
  pack_b   <<<1024,  256, 0, stream>>>(Wenc,  WencB,  1024, 256);
  pack_b   <<<256,   256, 0, stream>>>(Wpred, WpredB, 256,  256);
  pack_b   <<<256,   256, 0, stream>>>(Wact,  WactB,  256,  256);
  pack_x   <<<32768, 256, 0, stream>>>(x, xA);
  init_state<<<384,  256, 0, stream>>>(bih, bhh, biasg, zA, cstate);

  // sequential recurrence: stream ordering provides the inter-kernel deps
  for (int t = 0; t < SEQT; ++t) {
    const __bf16* xAt = xA + (size_t)t * 16384;
    gates_kernel   <<<64,  256, 0, stream>>>(xAt, zA, WcatB, biasg, gates);
    lstm_ew        <<<256, 256, 0, stream>>>(gates, cstate, zA);
    enc_kernel     <<<8,   256, 0, stream>>>(zA, WencB, benc, out, t);
    pred_act_kernel<<<16,  256, 0, stream>>>(zA, WpredB, WactB, bpred, bact, out, t);
  }
}
